// ResRnn_893353197832
// MI455X (gfx1250) — matrix operand order repaired and checked by reading, not compiled, not run on hardware
//
#include <hip/hip_runtime.h>
#include <hip/hip_bf16.h>

typedef __attribute__((ext_vector_type(16))) __bf16 v16bf;
typedef __attribute__((ext_vector_type(8)))  __bf16 v8bf;
typedef __attribute__((ext_vector_type(8)))  float  v8f;

#define S_LEN 2048
#define BATCH 64
#define IN_W  256
#define SW    512
#define OUT_W 128
#define BEND  1e-5f
#define NWG   32
#define NTHR  128

// ---------------------------------------------------------------------------
// Device-wide barrier: monotonic counter in L2, one atomic per workgroup per
// round. Counter is reset to 0 by the prep kernel each launch, so every call
// (and every graph replay) behaves identically.
// ---------------------------------------------------------------------------
__device__ __forceinline__ void grid_barrier(unsigned* cnt) {
    __syncthreads();
    if (threadIdx.x == 0) {
        __threadfence();
        unsigned old = __hip_atomic_fetch_add(cnt, 1u, __ATOMIC_ACQ_REL,
                                              __HIP_MEMORY_SCOPE_AGENT);
        unsigned target = old - (old % NWG) + NWG;
        while (__hip_atomic_load(cnt, __ATOMIC_ACQUIRE,
                                 __HIP_MEMORY_SCOPE_AGENT) < target) {
            __builtin_amdgcn_s_sleep(1);
        }
    }
    __syncthreads();
}

// ---------------------------------------------------------------------------
// Prep: reset barrier, build inverse permutation, build step-0 permuted state
//   sp0[b, j] = initial_stream[perm[j]]*BEND + x[0, b, perm[j]]   (x padded)
// ---------------------------------------------------------------------------
__global__ void res_prep(const float* __restrict__ input,
                         const float* __restrict__ istream,
                         const int*   __restrict__ perm,
                         float* __restrict__ sp32, __bf16* __restrict__ sp16,
                         int* __restrict__ ip, unsigned* __restrict__ cnt)
{
    const int i = blockIdx.x * blockDim.x + threadIdx.x;
    if (i == 0) *cnt = 0u;
    if (i < SW) ip[perm[i]] = i;
    if (i < BATCH * SW) {
        const int b = i >> 9;
        const int j = i & (SW - 1);
        const int c = perm[j];
        float v = istream[c] * BEND;
        if (c < IN_W) v += input[b * IN_W + c];
        sp32[i] = v;
        sp16[i] = (__bf16)v;
    }
}

// ---------------------------------------------------------------------------
// Persistent scan kernel. Each of the 32 WGs owns 16 output columns (n0).
// Each of its 4 waves owns one 16-row batch tile (m0). W1/W2 column slices
// are converted to bf16 and staged in LDS in WMMA B-fragment order ONCE.
// ---------------------------------------------------------------------------
__global__ __launch_bounds__(NTHR, 1)
void res_persist(const float* __restrict__ input,
                 const float* __restrict__ W1, const float* __restrict__ b1,
                 const float* __restrict__ W2, const float* __restrict__ b2,
                 const int* __restrict__ ip,
                 float* __restrict__ spA32, float* __restrict__ spB32,
                 __bf16* __restrict__ spA16, __bf16* __restrict__ spB16,
                 __bf16* __restrict__ h16,
                 float* __restrict__ out,
                 unsigned* __restrict__ cnt)
{
    // 16 K-chunks x 32 lanes x 16 bf16 = 16KB per weight slice
    __shared__ __attribute__((aligned(32))) __bf16 lsW1[16 * 32 * 16];
    __shared__ __attribute__((aligned(32))) __bf16 lsW2[16 * 32 * 16];

    const int n0 = blockIdx.x * 16;

    // Stage B-matrix fragments: B[k,n] = W[n0+n, k]. Per ISA, 16-bit B 32x16:
    // lane l holds column n = l&15, K = (e>>3)*16 + (l>>4)*8 + (e&7).
    for (int idx = threadIdx.x; idx < 16 * 32 * 16; idx += NTHR) {
        const int kb = idx >> 9;
        const int l  = (idx >> 4) & 31;
        const int e  = idx & 15;
        const int n  = l & 15;
        const int k  = kb * 32 + ((e >> 3) << 4) + ((l >> 4) << 3) + (e & 7);
        lsW1[idx] = (__bf16)W1[(n0 + n) * SW + k];
        lsW2[idx] = (__bf16)W2[(n0 + n) * SW + k];
    }
    __syncthreads();

    const int lane = threadIdx.x & 31;
    const int wv   = threadIdx.x >> 5;    // 0..3 -> M tile
    const int g    = lane >> 4;           // lane group
    const int col  = lane & 15;
    const int m0   = wv << 4;
    const int c    = n0 + col;            // this lane's output column
    const int jn   = ip[c];               // where column c lands after perm
    const float bias1 = b1[c];
    const float bias2 = b2[c];

    const float*  spR32 = spA32;  float*  spW32 = spB32;
    const __bf16* spR16 = spA16;  __bf16* spW16 = spB16;

    for (int t = 0; t < S_LEN; ++t) {
        // ---------- Phase A: h = relu(sp @ W1^T + b1), tile (m0, n0) ----------
        v8f acc;
        #pragma unroll
        for (int r = 0; r < 8; ++r) acc[r] = bias1;
        {
            // A fragment (16-bit 16x32): lane row m = m0 + (lane&15);
            // low 8 elems = K kb*32 + g*8.., high 8 = K kb*32+16+g*8..
            const __bf16* Arow = spR16 + (m0 + col) * SW;
            #pragma unroll
            for (int kb = 0; kb < 16; ++kb) {
                v8bf alo = *(const v8bf*)(Arow + kb * 32 + g * 8);
                v8bf ahi = *(const v8bf*)(Arow + kb * 32 + 16 + g * 8);
                v16bf a  = __builtin_shufflevector(alo, ahi,
                             0,1,2,3,4,5,6,7,8,9,10,11,12,13,14,15);
                v16bf bm = *(const v16bf*)(lsW1 + kb * 512 + lane * 16);
                acc = __builtin_amdgcn_wmma_f32_16x16x32_bf16(
                          false, a, false, bm, (short)0, acc, false, false);
            }
        }
        // C/D layout: element r -> row m0 + r + 8*g, col = lane&15
        #pragma unroll
        for (int r = 0; r < 8; ++r) {
            const int b = m0 + r + (g << 3);
            float hv = acc[r];
            hv = hv > 0.0f ? hv : 0.0f;
            h16[b * SW + c] = (__bf16)hv;
        }
        grid_barrier(cnt);

        // ---------- Phase B: u = h @ W2^T + b2 ; state update ----------
        v8f acc2;
        #pragma unroll
        for (int r = 0; r < 8; ++r) acc2[r] = bias2;
        {
            const __bf16* Arow = h16 + (m0 + col) * SW;
            #pragma unroll
            for (int kb = 0; kb < 16; ++kb) {
                v8bf alo = *(const v8bf*)(Arow + kb * 32 + g * 8);
                v8bf ahi = *(const v8bf*)(Arow + kb * 32 + 16 + g * 8);
                v16bf a  = __builtin_shufflevector(alo, ahi,
                             0,1,2,3,4,5,6,7,8,9,10,11,12,13,14,15);
                v16bf bm = *(const v16bf*)(lsW2 + kb * 512 + lane * 16);
                acc2 = __builtin_amdgcn_wmma_f32_16x16x32_bf16(
                           false, a, false, bm, (short)0, acc2, false, false);
            }
        }
        #pragma unroll
        for (int r = 0; r < 8; ++r) {
            const int b  = m0 + r + (g << 3);
            const float sv = spR32[b * SW + c] + BEND * acc2[r];  // carried state
            // Fold next step's "add x, then permute" into this scatter:
            // sp_next[b, invperm[c]] = sv + x[t+1, b, c]
            float xn = 0.0f;
            if (c < IN_W && (t + 1) < S_LEN) {
                xn = input[((long)(t + 1) * BATCH + b) * IN_W + c];
                if (r == 0 && (t + 2) < S_LEN)
                    __builtin_prefetch(
                        &input[((long)(t + 2) * BATCH + b) * IN_W + c], 0, 1);
            }
            const float nv = sv + xn;
            spW32[b * SW + jn] = nv;
            spW16[b * SW + jn] = (__bf16)nv;
            if (t == S_LEN - 1) {
                out[OUT_W * BATCH + b * SW + c] = sv;              // "last"
                if (c >= SW - OUT_W)
                    out[b * OUT_W + (c - (SW - OUT_W))] = sv;      // "outputs"
            }
        }
        grid_barrier(cnt);

        // ping-pong state buffers
        const float*  t32 = spR32; spR32 = spW32; spW32 = (float*)t32;
        const __bf16* t16 = spR16; spR16 = spW16; spW16 = (__bf16*)t16;
    }
}

// ---------------------------------------------------------------------------
extern "C" void kernel_launch(void* const* d_in, const int* in_sizes, int n_in,
                              void* d_out, int out_size, void* d_ws, size_t ws_size,
                              hipStream_t stream)
{
    const float* input   = (const float*)d_in[0];
    const float* istream = (const float*)d_in[1];
    const float* W1      = (const float*)d_in[2];
    const float* b1      = (const float*)d_in[3];
    const float* W2      = (const float*)d_in[4];
    const float* b2      = (const float*)d_in[5];
    const int*   perm    = (const int*)d_in[6];
    float* out = (float*)d_out;

    char* ws = (char*)d_ws;
    unsigned* cnt = (unsigned*)ws;                               // 4 B
    int*      ip  = (int*)(ws + 1024);                           // 2 KB
    float*  spA32 = (float*)(ws + 4096);                         // 128 KB
    float*  spB32 = (float*)(ws + 4096 + 131072);                // 128 KB
    __bf16* spA16 = (__bf16*)(ws + 4096 + 262144);               // 64 KB
    __bf16* spB16 = (__bf16*)(ws + 4096 + 262144 + 65536);       // 64 KB
    __bf16* h16   = (__bf16*)(ws + 4096 + 262144 + 131072);      // 64 KB

    res_prep<<<(BATCH * SW + 255) / 256, 256, 0, stream>>>(
        input, istream, perm, spA32, spA16, ip, cnt);

    res_persist<<<NWG, NTHR, 0, stream>>>(
        input, W1, b1, W2, b2, ip,
        spA32, spB32, spA16, spB16, h16, out, cnt);
}
